// GNNModel_12876311953727
// MI455X (gfx1250) — matrix operand order repaired and checked by reading, not compiled, not run on hardware
//
#include <hip/hip_runtime.h>

// Problem constants (from the reference)
#define NN   50000
#define NE   800000
#define H    128
#define EF   16
#define MSGK 144            // H + EF
#define NLAYERS 3

#define W1FRAG (5*8*512)    // 5 K-chunks (K padded to 160) x 8 col-tiles x 32 lanes x 16 halves
#define W2FRAG (4*8*512)    // 4 K-chunks x 8 col-tiles x 32 lanes x 16 halves
#define LPACK  (W1FRAG + 2*W2FRAG)   // packed f16 fragments per layer (W1 | W2 | Wself)
#define HSTR   136          // hidden-tile LDS row stride in halves (bank-conflict-free)

typedef __attribute__((ext_vector_type(16))) _Float16 v16h;
typedef __attribute__((ext_vector_type(8)))  _Float16 v8h;
typedef __attribute__((ext_vector_type(8)))  float    v8f;
typedef __attribute__((ext_vector_type(4)))  float    v4f;

union V16u { v16h v; v8h h[2]; };

static __device__ __forceinline__ v8f wmma_f16(v16h a, v16h b, v8f c) {
  // D = A(16x32 f16) * B(32x16 f16) + C(16x16 f32)
  return __builtin_amdgcn_wmma_f32_16x16x32_f16(false, a, false, b, (short)0, c,
                                                false, false);
}

// Build an A-fragment (one 16x32 f16 K-chunk) from two 8-float f32 runs (p0, p1) in global/L2.
// ISA 16-bit A layout: lane-half lh owns k = 8*(2*g + lh) + (0..7) for g=0,1, i.e.
// two contiguous 8-float runs at kbase and kbase+16 (kbase = chunk*32 + lh*8).
static __device__ __forceinline__ v16h frag_a_from_f32(const float* p0, const float* p1) {
  v4f q0 = ((const v4f*)p0)[0];
  v4f q1 = ((const v4f*)p0)[1];
  v4f q2 = ((const v4f*)p1)[0];
  v4f q3 = ((const v4f*)p1)[1];
  v16h a;
#pragma unroll
  for (int i = 0; i < 4; ++i) {
    a[i]      = (_Float16)q0[i];
    a[i + 4]  = (_Float16)q1[i];
    a[i + 8]  = (_Float16)q2[i];
    a[i + 12] = (_Float16)q3[i];
  }
  return a;
}

// Async-DMA `nbytes` (multiple of 16B, 16B aligned) from global `src` into LDS
// starting at LDS byte offset `lds_off`.  Uses GLOBAL_LOAD_ASYNC_TO_LDS_B128 in
// GVS mode (SADDR = 64-bit base SGPR pair, VADDR = 32-bit offset, VDST = LDS
// byte offset).  Tracked by ASYNCcnt; waits asynccnt 0 itself; caller must barrier.
static __device__ __forceinline__ void async_copy_to_lds(const _Float16* src,
                                                         unsigned lds_off,
                                                         int nbytes, int tid,
                                                         int nthreads) {
  const unsigned long long gbase = (unsigned long long)src;
  for (int idx = tid; idx < (nbytes >> 4); idx += nthreads) {
    const unsigned goff = (unsigned)(idx << 4);
    const unsigned loff = lds_off + goff;
    asm volatile("global_load_async_to_lds_b128 %0, %1, %2"
                 :: "v"(loff), "v"(goff), "s"(gbase) : "memory");
  }
  asm volatile("s_wait_asynccnt 0x0" ::: "memory");
}

// ---------------------------------------------------------------------------
// One-shot: pre-swizzle all layers' weights into fragment-native f16 blobs so
// per-WG LDS staging becomes a pure async DMA.  Fragment element (c,t,lane,i)
// holds W[k][n] with k = c*32 + (lane>>4)*8 + (i&7) + (i>>3)*16, n = t*16 + (lane&15).
// ---------------------------------------------------------------------------
__global__ void pack_weights_kernel(const float* __restrict__ W1,
                                    const float* __restrict__ W2,
                                    const float* __restrict__ Wself,
                                    _Float16* __restrict__ wpack) {
  const int gid = blockIdx.x * 256 + threadIdx.x;
  if (gid >= NLAYERS * LPACK) return;
  const int l   = gid / LPACK;
  const int idx = gid % LPACK;
  float v;
  if (idx < W1FRAG) {
    const int j = idx;
    const int i = j & 15, lane = (j >> 4) & 31, t = (j >> 9) & 7, c = j >> 12;
    const int k = c * 32 + (lane >> 4) * 8 + (i & 7) + (i >> 3) * 16;
    const int n = t * 16 + (lane & 15);
    v = (k < MSGK) ? W1[(size_t)l * MSGK * H + k * H + n] : 0.0f;   // zero-pad K 144..159
  } else if (idx < W1FRAG + W2FRAG) {
    const int j = idx - W1FRAG;
    const int i = j & 15, lane = (j >> 4) & 31, t = (j >> 9) & 7, c = j >> 12;
    const int k = c * 32 + (lane >> 4) * 8 + (i & 7) + (i >> 3) * 16;
    const int n = t * 16 + (lane & 15);
    v = W2[(size_t)l * H * H + k * H + n];
  } else {
    const int j = idx - W1FRAG - W2FRAG;
    const int i = j & 15, lane = (j >> 4) & 31, t = (j >> 9) & 7, c = j >> 12;
    const int k = c * 32 + (lane >> 4) * 8 + (i & 7) + (i >> 3) * 16;
    const int n = t * 16 + (lane & 15);
    v = Wself[(size_t)l * H * H + k * H + n];
  }
  wpack[(size_t)l * LPACK + idx] = (_Float16)v;
}

// ---------------------------------------------------------------------------
// Per-edge message MLP + scatter-add:  aggr[dst] += W2^T relu(W1^T [x_src; ea] + b1) + b2
// 128 threads = 4 waves; each wave does 4 tiles of 16 edges (256 edges / WG).
// wpk points at this layer's packed [W1FRAG | W2FRAG] f16 fragment blob.
// ---------------------------------------------------------------------------
__global__ __launch_bounds__(128)
void edge_mlp_kernel(const float* __restrict__ x,    const int* __restrict__ esrc,
                     const int* __restrict__ edst,   const float* __restrict__ eattr,
                     const _Float16* __restrict__ wpk,
                     const float* __restrict__ b1,   const float* __restrict__ b2,
                     float* __restrict__ aggr)
{
  extern __shared__ char smem[];          // dynamic LDS starts at offset 0 (no static __shared__)
  _Float16* w1f    = (_Float16*)smem;          // 40 KB: W1 fragments (K padded 144->160)
  _Float16* w2f    = w1f + W1FRAG;             // 32 KB: W2 fragments
  _Float16* hidAll = w2f + W2FRAG;             // 4 waves x 16 x HSTR halves
  int*      dstb   = (int*)(hidAll + 4 * 16 * HSTR);

  const int tid = threadIdx.x;

  // DMA both weight blobs into LDS (byte-identical layout), then barrier.
  async_copy_to_lds(wpk, 0u, (W1FRAG + W2FRAG) * (int)sizeof(_Float16), tid, 128);
  __syncthreads();

  const int wave = tid >> 5, lane = tid & 31;
  const int ln = lane & 15, lh = lane >> 4;
  _Float16* hid  = hidAll + wave * 16 * HSTR;
  int*      dstw = dstb + wave * 16;

#pragma unroll 1
  for (int tt = 0; tt < 4; ++tt) {
    const int ebase = (blockIdx.x * 4 + wave) * 64 + tt * 16;
    const int e = ebase + ln;                    // lanes L and L+16 share one edge row
    if (lane < 16) dstw[ln] = edst[e];

    // A fragments straight from global (gather of x[src] + edge_attr, zero-pad K 144..159)
    const int src = esrc[e];
    const float* xr = x + (size_t)src * H;
    v16h A[5];
#pragma unroll
    for (int c = 0; c < 4; ++c) {
      const float* p = xr + c * 32 + lh * 8;
      A[c] = frag_a_from_f32(p, p + 16);
    }
    {
      const float* ea = eattr + (size_t)e * EF + lh * 8;
      v16h a;
#pragma unroll
      for (int i = 0; i < 8; ++i) { a[i] = (_Float16)ea[i]; a[i + 8] = (_Float16)0.0f; }
      A[4] = a;
    }
    if (tt < 3) {                                 // prefetch next tile's gathered row
      const int sn = esrc[e + 16];
      __builtin_prefetch(x + (size_t)sn * H + lh * 64, 0, 0);
    }

    // GEMM1: hidden(16x128) = relu(A(16x160) * W1 + b1), stored as f16 in LDS
#pragma unroll
    for (int t = 0; t < 8; ++t) {
      const float bv = b1[t * 16 + ln];
      v8f acc;
#pragma unroll
      for (int r = 0; r < 8; ++r) acc[r] = bv;
#pragma unroll
      for (int c = 0; c < 5; ++c) {
        const v16h B = *(const v16h*)(w1f + ((c * 8 + t) * 32 + lane) * 16);
        acc = wmma_f16(A[c], B, acc);
      }
#pragma unroll
      for (int r = 0; r < 8; ++r) {
        const float v = fmaxf(acc[r], 0.0f);
        hid[(r + 8 * lh) * HSTR + t * 16 + ln] = (_Float16)v;   // C-layout -> row-major f16
      }
    }
    __syncthreads();   // uniform across WG; orders hid writes vs A2 reads

    // GEMM2: h(16x128) = hidden * W2 + b2 ; then scatter-add into aggr[dst]
    v16h A2[4];
#pragma unroll
    for (int c = 0; c < 4; ++c) {
      const _Float16* p = hid + ln * HSTR + c * 32 + lh * 8;
      V16u u; u.h[0] = *(const v8h*)p; u.h[1] = *(const v8h*)(p + 16);
      A2[c] = u.v;
    }
    int didx[8];
#pragma unroll
    for (int r = 0; r < 8; ++r) didx[r] = dstw[r + 8 * lh];

#pragma unroll
    for (int t = 0; t < 8; ++t) {
      const float bv = b2[t * 16 + ln];
      v8f acc;
#pragma unroll
      for (int r = 0; r < 8; ++r) acc[r] = bv;
#pragma unroll
      for (int c = 0; c < 4; ++c) {
        const v16h B = *(const v16h*)(w2f + ((c * 8 + t) * 32 + lane) * 16);
        acc = wmma_f16(A2[c], B, acc);
      }
#pragma unroll
      for (int r = 0; r < 8; ++r)
        atomicAdd(aggr + (size_t)didx[r] * H + t * 16 + ln, acc[r]);  // segment_sum
    }
    __syncthreads();
  }
}

// ---------------------------------------------------------------------------
// x_out = x_in @ Wself + bself + aggr * invdeg     (in-place safe per-tile)
// wpk points at this layer's packed Wself fragment blob.
// ---------------------------------------------------------------------------
__global__ __launch_bounds__(128)
void node_update_kernel(const float* __restrict__ xin,
                        const _Float16* __restrict__ wpk,
                        const float* __restrict__ bs,   const float* __restrict__ aggr,
                        const float* __restrict__ invdeg, float* __restrict__ xout)
{
  extern __shared__ char smem[];
  _Float16* wf = (_Float16*)smem;              // 32 KB Wself fragments

  const int tid = threadIdx.x;
  async_copy_to_lds(wpk, 0u, W2FRAG * (int)sizeof(_Float16), tid, 128);
  __syncthreads();

  const int wave = tid >> 5, lane = tid & 31;
  const int ln = lane & 15, lh = lane >> 4;

#pragma unroll 1
  for (int tt = 0; tt < 4; ++tt) {
    const int nbase = (blockIdx.x * 4 + wave) * 64 + tt * 16;
    const int row = nbase + ln;
    const int rc = row < NN ? row : NN - 1;      // clamp reads; EXEC stays all-1 for WMMA
    const float* xr = xin + (size_t)rc * H;

    v16h A[4];
#pragma unroll
    for (int c = 0; c < 4; ++c) {
      const float* p = xr + c * 32 + lh * 8;
      A[c] = frag_a_from_f32(p, p + 16);
    }

    int   rm[8];
    float idg[8];
#pragma unroll
    for (int r = 0; r < 8; ++r) {
      rm[r]  = nbase + r + 8 * lh;
      idg[r] = (rm[r] < NN) ? invdeg[rm[r]] : 0.0f;
    }

#pragma unroll
    for (int t = 0; t < 8; ++t) {
      const float bv = bs[t * 16 + ln];
      v8f acc;
#pragma unroll
      for (int r = 0; r < 8; ++r) acc[r] = bv;
#pragma unroll
      for (int c = 0; c < 4; ++c) {
        const v16h B = *(const v16h*)(wf + ((c * 8 + t) * 32 + lane) * 16);
        acc = wmma_f16(A[c], B, acc);
      }
      const int col = t * 16 + ln;
#pragma unroll
      for (int r = 0; r < 8; ++r) {
        if (rm[r] < NN) {
          const size_t o = (size_t)rm[r] * H + col;
          xout[o] = acc[r] + aggr[o] * idg[r];
        }
      }
    }
  }
}

// ---------------------------------------------------------------------------
// Small helpers: zero fill, in-degree count, reciprocal of max(cnt,1)
// ---------------------------------------------------------------------------
__global__ void zero_kernel(float* __restrict__ p, int n) {
  const int i = blockIdx.x * 256 + threadIdx.x;
  if (i < n) p[i] = 0.0f;
}
__global__ void degree_kernel(const int* __restrict__ edst, float* __restrict__ cnt) {
  const int e = blockIdx.x * 256 + threadIdx.x;
  if (e < NE) atomicAdd(cnt + edst[e], 1.0f);
}
__global__ void invdeg_kernel(const float* __restrict__ cnt, float* __restrict__ inv) {
  const int i = blockIdx.x * 256 + threadIdx.x;
  if (i < NN) inv[i] = 1.0f / fmaxf(cnt[i], 1.0f);
}

// ---------------------------------------------------------------------------
// Orchestration.  Workspace: aggr (N*H f32, aliased as degree counter before
// first use) | xbuf (N*H f32) | invdeg (N f32) | wpack (3*LPACK f16) ~ 51.7 MB.
// Layer flow: d_in -> xbuf, xbuf -> xbuf (in-place), xbuf -> d_out.
// ---------------------------------------------------------------------------
extern "C" void kernel_launch(void* const* d_in, const int* in_sizes, int n_in,
                              void* d_out, int out_size, void* d_ws, size_t ws_size,
                              hipStream_t stream)
{
  (void)in_sizes; (void)n_in; (void)out_size; (void)ws_size;

  const float* x     = (const float*)d_in[0];
  const int*   ei    = (const int*)  d_in[1];
  const float* eattr = (const float*)d_in[2];
  const float* W1    = (const float*)d_in[3];
  const float* b1    = (const float*)d_in[4];
  const float* W2    = (const float*)d_in[5];
  const float* b2    = (const float*)d_in[6];
  const float* Wself = (const float*)d_in[7];
  const float* bself = (const float*)d_in[8];
  float*       out   = (float*)d_out;

  float*    aggr   = (float*)d_ws;
  float*    xbuf   = aggr + (size_t)NN * H;
  float*    invdeg = xbuf + (size_t)NN * H;
  _Float16* wpack  = (_Float16*)(invdeg + NN);    // 16B-aligned (offset 51,400,000 B)
  float*    cnt    = aggr;                        // alias: used before aggr is first zeroed

  const int* esrc = ei;
  const int* edst = ei + NE;

  pack_weights_kernel<<<(NLAYERS * LPACK + 255) / 256, 256, 0, stream>>>(W1, W2, Wself, wpack);
  zero_kernel  <<<(NN + 255) / 256, 256, 0, stream>>>(cnt, NN);
  degree_kernel<<<(NE + 255) / 256, 256, 0, stream>>>(edst, cnt);
  invdeg_kernel<<<(NN + 255) / 256, 256, 0, stream>>>(cnt, invdeg);

  const size_t eshm = (size_t)(W1FRAG + W2FRAG + 4 * 16 * HSTR) * sizeof(_Float16)
                    + 64 * sizeof(int);                      // 91392 B (< 320 KB/WG)
  const size_t nshm = (size_t)W2FRAG * sizeof(_Float16);     // 32768 B

  for (int l = 0; l < NLAYERS; ++l) {
    const float* xin  = (l == 0) ? x   : xbuf;
    float*       xout = (l == 2) ? out : xbuf;

    zero_kernel<<<(NN * H + 255) / 256, 256, 0, stream>>>(aggr, NN * H);

    edge_mlp_kernel<<<NE / 256, 128, eshm, stream>>>(
        xin, esrc, edst, eattr,
        wpack + (size_t)l * LPACK,
        b1 + (size_t)l * H, b2 + (size_t)l * H, aggr);

    node_update_kernel<<<(NN + 255) / 256, 128, nshm, stream>>>(
        xin, wpack + (size_t)l * LPACK + W1FRAG + W2FRAG,
        bself + (size_t)l * H, aggr, invdeg, xout);
  }
}
